// ImportanceRenderer_77670188581442
// MI455X (gfx1250) — compile-verified, matrix operand order checked by reading, not run on hardware
//
#include <hip/hip_runtime.h>
#include <hip/hip_bf16.h>
#include <math.h>

// ---------------------------------------------------------------------------
// EG3D-style tri-plane importance renderer for gfx1250 (MI455X, wave32).
// MLP (32->64->33) runs on v_wmma_f32_16x16x32_f16 (16-sample tiles, one tile
// per wave). Weights are pre-swizzled into fragment-native LDS layout so each
// B fragment is a single aligned 32-byte LDS read (2x ds_load_b128).
// Transcendentals use the hardware fast paths (v_exp_f32 / v_log_f32 / rcp) —
// consistent with f16 WMMA numerics, removes libm range-reduction bloat.
// ---------------------------------------------------------------------------

typedef __attribute__((ext_vector_type(16))) _Float16 v16h;
typedef __attribute__((ext_vector_type(8)))  float    v8f;

#define BB   2
#define RR   4096
#define SC   48
#define SF   48
#define CP   32      // plane channels
#define HPX  256     // plane resolution
#define HID  64      // hidden width
#define NO   33      // 1 sigma + 32 rgb
#define NOP  48      // padded output width (3 N-tiles of 16)
#define NRAY (BB*RR)
#define NSAMP (NRAY*SC)

#define RAY_START 2.25f
#define RAY_END   3.30f
#define DELTA_C   ((RAY_END - RAY_START) / (SC - 1))

// Hardware-native fast transcendentals (single v_exp_f32 / v_log_f32 / v_rcp).
__device__ __forceinline__ float fast_exp(float x)  { return __expf(x); }
__device__ __forceinline__ float softplusf(float x) {
    return (x > 20.f) ? x : __logf(1.f + __expf(x));
}
__device__ __forceinline__ float sigmoidf(float x) {
    return __fdividef(1.f, 1.f + __expf(-x));
}
__device__ __forceinline__ float coarse_depth(int s) {
    return RAY_START + ((float)s + 0.5f) * DELTA_C;
}

// ---------------------------------------------------------------------------
// Fused tri-plane sampling + MLP pass. One wave = one tile of 16 consecutive
// samples of one ray (48 samples = 3 tiles). 8 waves/block, exact grid, so
// EXEC is all-1s at every WMMA site.
// ---------------------------------------------------------------------------
__global__ void __launch_bounds__(256)
nerf_mlp_kernel(const float* __restrict__ planes,
                const float* __restrict__ ray_o,
                const float* __restrict__ ray_d,
                const float* __restrict__ w1, const float* __restrict__ b1,
                const float* __restrict__ w2, const float* __restrict__ b2,
                const float* __restrict__ depths,
                float* __restrict__ sigma_out,      // [NSAMP]
                float* __restrict__ rgb_out,        // [NSAMP*32]
                float* __restrict__ sdf_out,        // may be null
                int use_analytic)
{
    // Fragment-native weight staging: each [..][lane] entry is the v16h B
    // fragment held by that lane (col = lane&15 within N-tile, K-half = lane>>4).
    __shared__ v16h  sW1f[4][32];        // layer1: 4 N-tiles          (4 KB)
    __shared__ v16h  sW2f[2][3][32];     // layer2: 2 K-halves x 3 N   (6 KB)
    __shared__ float sB1[HID];
    __shared__ float sB2[NOP];
    __shared__ v16h  sHf[8][16][4];      // per-wave h: [m][k-quarter] (16 KB)

    {   // ---- cooperative swizzled staging (f32 -> f16) ----
        _Float16* w1f = (_Float16*)&sW1f[0][0];
        for (int idx = threadIdx.x; idx < 4 * 32 * 16; idx += 256) {
            const int j  = idx & 15;
            const int ln = (idx >> 4) & 31;
            const int nt = idx >> 9;
            const int k  = ((ln >> 4) << 4) + j;
            const int n  = nt * 16 + (ln & 15);
            w1f[idx] = (_Float16)w1[k * HID + n];
        }
        _Float16* w2f = (_Float16*)&sW2f[0][0][0];
        for (int idx = threadIdx.x; idx < 2 * 3 * 32 * 16; idx += 256) {
            const int j  = idx & 15;
            const int ln = (idx >> 4) & 31;
            const int nt = (idx >> 9) % 3;
            const int kh = (idx >> 9) / 3;
            const int k  = kh * 32 + ((ln >> 4) << 4) + j;
            const int n  = nt * 16 + (ln & 15);
            w2f[idx] = (_Float16)((n < NO) ? w2[k * NO + n] : 0.f);
        }
        if (threadIdx.x < HID) sB1[threadIdx.x] = b1[threadIdx.x];
        if (threadIdx.x < NOP)
            sB2[threadIdx.x] = (threadIdx.x < NO) ? b2[threadIdx.x] : 0.f;
    }
    __syncthreads();

    const int wave  = threadIdx.x >> 5;
    const int lane  = threadIdx.x & 31;
    const int mrow  = lane & 15;
    const int khalf = lane >> 4;

    const int tile  = blockIdx.x * 8 + wave;
    const int ray   = tile / 3;
    const int sbase = (tile % 3) * 16;
    const int bidx  = ray / RR;
    const int s     = sbase + mrow;
    const int gsamp = ray * SC + s;

    // ---- sample position ----
    const float dpt = use_analytic ? coarse_depth(s) : depths[gsamp];
    const float ox = ray_o[ray * 3 + 0], oy = ray_o[ray * 3 + 1], oz = ray_o[ray * 3 + 2];
    const float dx = ray_d[ray * 3 + 0], dy = ray_d[ray * 3 + 1], dz = ray_d[ray * 3 + 2];
    const float cx = ox + dpt * dx;     // (2/BOX_WARP) == 1
    const float cy = oy + dpt * dy;
    const float cz = oz + dpt * dz;

    // ---- tri-plane bilinear gather, branch-free (clamp + zero weight) ----
    float feat[16];
#pragma unroll
    for (int j = 0; j < 16; ++j) feat[j] = 0.f;

#pragma unroll
    for (int p = 0; p < 3; ++p) {
        // plane0:(x,y)  plane1:(x,z)  plane2:(z,x)   (from INV_PLANES)
        const float u = (p == 0) ? cx : (p == 1) ? cx : cz;
        const float v = (p == 0) ? cy : (p == 1) ? cz : cx;
        const float fx = ((u + 1.f) * (float)HPX - 1.f) * 0.5f;
        const float fy = ((v + 1.f) * (float)HPX - 1.f) * 0.5f;
        const float x0f = floorf(fx), y0f = floorf(fy);
        const float wx = fx - x0f, wy = fy - y0f;
        const int x0 = (int)x0f, y0 = (int)y0f;
        const int x1 = x0 + 1,   y1 = y0 + 1;

        const float vx0 = (x0 >= 0 && x0 < HPX) ? 1.f : 0.f;
        const float vx1 = (x1 >= 0 && x1 < HPX) ? 1.f : 0.f;
        const float vy0 = (y0 >= 0 && y0 < HPX) ? 1.f : 0.f;
        const float vy1 = (y1 >= 0 && y1 < HPX) ? 1.f : 0.f;
        const int x0c = min(max(x0, 0), HPX - 1), x1c = min(max(x1, 0), HPX - 1);
        const int y0c = min(max(y0, 0), HPX - 1), y1c = min(max(y1, 0), HPX - 1);

        const float w00 = (1.f - wx) * (1.f - wy) * vx0 * vy0;
        const float w01 = wx * (1.f - wy) * vx1 * vy0;
        const float w10 = (1.f - wx) * wy * vx0 * vy1;
        const float w11 = wx * wy * vx1 * vy1;
        const size_t o00 = (size_t)y0c * HPX + x0c;
        const size_t o01 = (size_t)y0c * HPX + x1c;
        const size_t o10 = (size_t)y1c * HPX + x0c;
        const size_t o11 = (size_t)y1c * HPX + x1c;

        const float* base =
            planes + (((size_t)bidx * 3 + p) * CP + khalf * 16) * (size_t)(HPX * HPX);
#pragma unroll
        for (int j = 0; j < 16; ++j) {
            const float* ch = base + (size_t)j * (HPX * HPX);
            feat[j] += w00 * ch[o00] + w01 * ch[o01] + w10 * ch[o10] + w11 * ch[o11];
        }
    }

    // A fragment (16x32 f16): lane m holds K 0..15 of row m, lane m+16 K 16..31
    v16h afrag;
#pragma unroll
    for (int j = 0; j < 16; ++j) afrag[j] = (_Float16)(feat[j] * (1.f / 3.f));

    // ---- layer 1: 16x32 @ 32x64 via 4 WMMAs ----
    v8f h1[4];
#pragma unroll
    for (int nt = 0; nt < 4; ++nt) {
        const v16h bf = sW1f[nt][lane];
        v8f c;
        const float bias = sB1[nt * 16 + mrow];
#pragma unroll
        for (int r = 0; r < 8; ++r) c[r] = bias;
        h1[nt] = __builtin_amdgcn_wmma_f32_16x16x32_f16(
            false, afrag, false, bf, (short)0, c, false, false);
    }

    // ---- softplus, transpose C-layout -> A-layout through LDS ----
    _Float16* hbuf = (_Float16*)&sHf[wave][0][0];
#pragma unroll
    for (int nt = 0; nt < 4; ++nt)
#pragma unroll
        for (int r = 0; r < 8; ++r) {
            const int row = r + 8 * khalf;
            const int col = nt * 16 + mrow;
            hbuf[row * HID + col] = (_Float16)softplusf(h1[nt][r]);
        }
    __syncthreads();   // uniform: every wave runs exactly one tile

    const v16h a20 = sHf[wave][mrow][khalf];       // K 0..31 half
    const v16h a21 = sHf[wave][mrow][2 + khalf];   // K 32..63 half

    // ---- layer 2: 16x64 @ 64x48 via 2(K) x 3(N) WMMAs ----
    v8f o2[3];
#pragma unroll
    for (int nt = 0; nt < 3; ++nt) {
        const v16h b0  = sW2f[0][nt][lane];
        const v16h b1f = sW2f[1][nt][lane];
        v8f c;
        const float bias = sB2[nt * 16 + mrow];
#pragma unroll
        for (int r = 0; r < 8; ++r) c[r] = bias;
        c = __builtin_amdgcn_wmma_f32_16x16x32_f16(
            false, a20, false, b0, (short)0, c, false, false);
        c = __builtin_amdgcn_wmma_f32_16x16x32_f16(
            false, a21, false, b1f, (short)0, c, false, false);
        o2[nt] = c;
    }

    // ---- scatter outputs: n==0 -> sigma, n 1..32 -> rgb channel n-1 ----
    if (mrow == 0) {
#pragma unroll
        for (int r = 0; r < 8; ++r) {
            const int row = r + 8 * khalf;
            const int g = ray * SC + sbase + row;
            sigma_out[g] = o2[0][r];
            if (sdf_out) sdf_out[g] = o2[0][r];
        }
    }
#pragma unroll
    for (int nt = 0; nt < 3; ++nt) {
        const int n = nt * 16 + mrow;
        if (n >= 1 && n <= 32) {
            const int ch = n - 1;
#pragma unroll
            for (int r = 0; r < 8; ++r) {
                const int row = r + 8 * khalf;
                const int g = ray * SC + sbase + row;
                rgb_out[(size_t)g * 32 + ch] =
                    sigmoidf(o2[nt][r]) * 1.002f - 0.001f;
            }
        }
    }
}

// ---------------------------------------------------------------------------
// Per-ray importance resampling: coarse ray-march weights -> maxpool smooth
// -> CDF -> inverse-transform (searchsorted) -> 48 sorted fine depths.
// ---------------------------------------------------------------------------
__global__ void importance_kernel(const float* __restrict__ sig_c,
                                  float* __restrict__ d_fine)
{
    const int ray = blockIdx.x * blockDim.x + threadIdx.x;
    if (ray >= NRAY) return;
    const float* sg = sig_c + (size_t)ray * SC;

    float wts[SC - 1];
    float T = 1.f;
#pragma unroll 1
    for (int i = 0; i < SC - 1; ++i) {
        const float dens  = softplusf(0.5f * (sg[i] + sg[i + 1]) - 1.f);
        const float alpha = 1.f - fast_exp(-dens * DELTA_C);
        wts[i] = alpha * T;
        T *= (1.f - alpha + 1e-10f);
    }
    float wmax[SC];
    for (int i = 0; i < SC; ++i) {
        const float a = (i >= 1 && i <= SC - 1) ? wts[i - 1] : 0.f;
        const float b = (i + 1 >= 1 && i + 1 <= SC - 1) ? wts[i] : 0.f;
        wmax[i] = fmaxf(a, b);
    }
    float pw[SC - 3];
    float total = 0.f;
    for (int i = 0; i < SC - 3; ++i) {
        pw[i] = 0.5f * (wmax[i + 1] + wmax[i + 2]) + 0.01f;
        total += pw[i];
    }
    float cdf[SC - 2];
    cdf[0] = 0.f;
    const float inv_total = __frcp_rn(total);
    for (int i = 0; i < SC - 3; ++i) cdf[i + 1] = cdf[i] + pw[i] * inv_total;

#pragma unroll 1
    for (int k = 0; k < SF; ++k) {
        const float u = (float)k / (float)(SF - 1);
        int ind = SC - 2;
        for (int i = 0; i < SC - 2; ++i) {
            if (cdf[i] > u) { ind = i; break; }
        }
        int below = ind - 1; if (below < 0) below = 0; if (below > SC - 3) below = SC - 3;
        int above = ind;     if (above > SC - 3) above = SC - 3;
        const float cb = cdf[below], ca = cdf[above];
        const float bb = RAY_START + (float)(below + 1) * DELTA_C;
        const float ba = RAY_START + (float)(above + 1) * DELTA_C;
        const float denom = (ca - cb < 1e-5f) ? 1.f : (ca - cb);
        d_fine[(size_t)ray * SF + k] = bb + __fdividef(u - cb, denom) * (ba - bb);
    }
}

// ---------------------------------------------------------------------------
// Merge (sorted coarse + sorted fine) + final ray march. One wave per ray,
// lane == rgb channel. Merge condition is lane-uniform.
// ---------------------------------------------------------------------------
__global__ void __launch_bounds__(256)
composite_kernel(const float* __restrict__ rgb_c, const float* __restrict__ sig_c,
                 const float* __restrict__ rgb_f, const float* __restrict__ sig_f,
                 const float* __restrict__ d_fine,
                 float* __restrict__ out_rgb, float* __restrict__ out_depth,
                 float* __restrict__ out_wsum)
{
    const int wave = threadIdx.x >> 5;
    const int lane = threadIdx.x & 31;
    const int ray  = blockIdx.x * 8 + wave;

    float T = 1.f, acc_c = 0.f, acc_d = 0.f, acc_w = 0.f;
    float pd = 0.f, ps = 0.f, pc = 0.f;
    int i = 0, j = 0;
#pragma unroll 1
    for (int k = 0; k < SC + SF; ++k) {
        const float dci = (i < SC) ? coarse_depth(i) : 3.0e38f;
        const float dfj = (j < SF) ? d_fine[(size_t)ray * SF + j] : 3.0e38f;
        float d, sgv, c;
        if (dci <= dfj) {   // ties: coarse first (stable argsort of concat)
            const int g = ray * SC + i;
            d = dci; sgv = sig_c[g]; c = rgb_c[(size_t)g * 32 + lane]; ++i;
        } else {
            const int g = ray * SF + j;
            d = dfj; sgv = sig_f[g]; c = rgb_f[(size_t)g * 32 + lane]; ++j;
        }
        if (k > 0) {
            const float delta = d - pd;
            const float dens  = softplusf(0.5f * (sgv + ps) - 1.f);
            const float alpha = 1.f - fast_exp(-dens * delta);
            const float w = alpha * T;
            acc_c += w * 0.5f * (c + pc);
            acc_d += w * 0.5f * (d + pd);
            acc_w += w;
            T *= (1.f - alpha + 1e-10f);
        }
        pd = d; ps = sgv; pc = c;
    }
    out_rgb[(size_t)ray * 32 + lane] = acc_c * 2.f - 1.f;
    if (lane == 0) { out_depth[ray] = acc_d; out_wsum[ray] = acc_w; }
}

// ---------------------------------------------------------------------------
extern "C" void kernel_launch(void* const* d_in, const int* in_sizes, int n_in,
                              void* d_out, int out_size, void* d_ws, size_t ws_size,
                              hipStream_t stream)
{
    const float* planes = (const float*)d_in[0];
    const float* ray_o  = (const float*)d_in[1];
    const float* ray_d  = (const float*)d_in[2];
    const float* w1     = (const float*)d_in[3];
    const float* b1     = (const float*)d_in[4];
    const float* w2     = (const float*)d_in[5];
    const float* b2     = (const float*)d_in[6];

    // d_out: rgb(B,R,32) | depth(B,R) | wsum(B,R) | sdf_coarse(B,R*48)
    float* out       = (float*)d_out;
    float* out_rgb   = out;
    float* out_depth = out_rgb + (size_t)NRAY * 32;
    float* out_wsum  = out_depth + NRAY;
    float* out_sdf   = out_wsum + NRAY;

    // workspace: sig_c | rgb_c | d_fine | sig_f | rgb_f   (~105 MB f32)
    float* ws     = (float*)d_ws;
    float* sig_c  = ws;
    float* rgb_c  = sig_c + NSAMP;
    float* dfine  = rgb_c + (size_t)NSAMP * 32;
    float* sig_f  = dfine + NSAMP;
    float* rgb_f  = sig_f + NSAMP;

    const int mlp_blocks = (NRAY * 3) / 8;   // 3072

    nerf_mlp_kernel<<<mlp_blocks, 256, 0, stream>>>(
        planes, ray_o, ray_d, w1, b1, w2, b2,
        nullptr, sig_c, rgb_c, out_sdf, /*use_analytic=*/1);

    importance_kernel<<<(NRAY + 255) / 256, 256, 0, stream>>>(sig_c, dfine);

    nerf_mlp_kernel<<<mlp_blocks, 256, 0, stream>>>(
        planes, ray_o, ray_d, w1, b1, w2, b2,
        dfine, sig_f, rgb_f, nullptr, /*use_analytic=*/0);

    composite_kernel<<<NRAY / 8, 256, 0, stream>>>(
        rgb_c, sig_c, rgb_f, sig_f, dfine, out_rgb, out_depth, out_wsum);
}
